// hmcf_14920716386881
// MI455X (gfx1250) — compile-verified, matrix operand order checked by reading, not run on hardware
//
#include <hip/hip_runtime.h>
#include <hip/hip_bf16.h>
#include <math.h>

#define N_USERS 100000
#define N_ITEMS 200000
#define N_NODES (N_USERS + N_ITEMS)
#define EMB_DIM 64
#define N_EDGES 4000000
#define N_LAYERS 2
#define BATCH 8192
#define EMB_REG 2.5e-05f

typedef __attribute__((ext_vector_type(16))) _Float16 v16h;
typedef __attribute__((ext_vector_type(8)))  float    v8f;

// ---------------------------------------------------------------- utilities
__global__ void k_zero(float* __restrict__ deg, float* __restrict__ loss2) {
    int i = blockIdx.x * blockDim.x + threadIdx.x;
    if (i < N_NODES) deg[i] = 0.0f;
    if (i < 2) loss2[i] = 0.0f;
}

// emb0 = concat(user_emb, item_emb); acc = emb0   (float4 granularity)
__global__ void k_init(const float* __restrict__ ue, const float* __restrict__ ie,
                       float4* __restrict__ bufA, float4* __restrict__ acc) {
    int i = blockIdx.x * blockDim.x + threadIdx.x;      // over N_NODES*16
    if (i >= N_NODES * 16) return;
    int node = i >> 4;
    int c    = i & 15;
    float4 v;
    if (node < N_USERS) v = ((const float4*)ue)[(size_t)node * 16 + c];
    else                v = ((const float4*)ie)[(size_t)(node - N_USERS) * 16 + c];
    bufA[i] = v;
    acc[i]  = v;
}

__global__ void k_deg(const int* __restrict__ h, float* __restrict__ deg) {
    int e = blockIdx.x * blockDim.x + threadIdx.x;
    if (e < N_EDGES) atomicAdd(&deg[h[e]], 1.0f);
}

__global__ void k_dinv(const float* __restrict__ deg, float* __restrict__ dinv) {
    int i = blockIdx.x * blockDim.x + threadIdx.x;
    if (i < N_NODES) {
        float d = deg[i];
        dinv[i] = (d > 0.0f) ? (1.0f / sqrtf(d)) : 0.0f;
    }
}

__global__ void k_copy(const float4* __restrict__ src, float4* __restrict__ dst, int n4) {
    int i = blockIdx.x * blockDim.x + threadIdx.x;
    if (i < n4) dst[i] = src[i];
}

__global__ void k_accum(float4* __restrict__ acc, const float4* __restrict__ nxt, int n4) {
    int i = blockIdx.x * blockDim.x + threadIdx.x;
    if (i < n4) {
        float4 a = acc[i];
        float4 b = nxt[i];
        a.x += b.x; a.y += b.y; a.z += b.z; a.w += b.w;
        acc[i] = a;
    }
}

// ------------------------------------------------------------------- SpMM
// nxt[h] += dinv[h]*dinv[t] * cur[t]; 16 threads per edge, float4 per thread
// (row reads and atomic scatters are 256B fully-coalesced per 16-lane group)
__global__ void k_spmm(const int* __restrict__ h, const int* __restrict__ t,
                       const float* __restrict__ dinv,
                       const float* __restrict__ cur, float* __restrict__ nxt) {
    long i = (long)blockIdx.x * blockDim.x + threadIdx.x;   // N_EDGES*16 work items
    if (i >= (long)N_EDGES * 16) return;
    int e = (int)(i >> 4);
    int c = (int)(i & 15) << 2;
    int hn = h[e];
    int tn = t[e];
    float w = dinv[hn] * dinv[tn];
    const float4 v = *(const float4*)&cur[(size_t)tn * EMB_DIM + c];
    float* dst = &nxt[(size_t)hn * EMB_DIM + c];
    atomicAdd(dst + 0, w * v.x);
    atomicAdd(dst + 1, w * v.y);
    atomicAdd(dst + 2, w * v.z);
    atomicAdd(dst + 3, w * v.w);
}

// ------------------------------------------------- BPR scoring via WMMA
// One wave per 16-row tile. scores = (U .* P) x ones(64x16) via
// v_wmma_f32_16x16x32_f16, K accumulated over two 32-chunks in f32.
// A layout (16-bit A 16x32): lane l (row r=l&15, hi=l>>4):
//   a[0..7]  = K = kc + hi*8 + 0..7
//   a[8..15] = K = kc + 16 + hi*8 + 0..7
// D layout: VGPR m = row m (lanes 0-15) / row m+8 (lanes 16-31); all columns
// equal since B is all ones -> lane 0 reads rows 0..7, lane 16 rows 8..15.
__global__ void k_mf_loss(const float* __restrict__ acc,
                          const int* __restrict__ users,
                          const int* __restrict__ pos_items,
                          const int* __restrict__ neg_items,
                          float* __restrict__ mf_accum) {
    const int lane = threadIdx.x & 31;
    const int tile = blockIdx.x * (blockDim.x >> 5) + (threadIdx.x >> 5);
    const int r  = lane & 15;
    const int hi = lane >> 4;
    const int b  = tile * 16 + r;                     // BATCH % 16 == 0, no tail

    const int un = users[b];
    const int pn = N_USERS + pos_items[b];
    const int nn = N_USERS + neg_items[b];
    const float* urow = acc + (size_t)un * EMB_DIM;
    const float* prow = acc + (size_t)pn * EMB_DIM;
    const float* nrow = acc + (size_t)nn * EMB_DIM;

    v16h bones;
#pragma unroll
    for (int i = 0; i < 16; ++i) bones[i] = (_Float16)1.0f;

    v8f cpos = {};
    v8f cneg = {};
#pragma unroll
    for (int kc = 0; kc < EMB_DIM; kc += 32) {
        v16h apos, aneg;
#pragma unroll
        for (int half = 0; half < 2; ++half) {
            const int kbase = kc + half * 16 + hi * 8;
#pragma unroll
            for (int j = 0; j < 8; ++j) {
                float uu = urow[kbase + j];
                apos[half * 8 + j] = (_Float16)(uu * prow[kbase + j]);
                aneg[half * 8 + j] = (_Float16)(uu * nrow[kbase + j]);
            }
        }
        cpos = __builtin_amdgcn_wmma_f32_16x16x32_f16(
            false, apos, false, bones, (short)0, cpos, false, false);
        cneg = __builtin_amdgcn_wmma_f32_16x16x32_f16(
            false, aneg, false, bones, (short)0, cneg, false, false);
    }

    if ((lane & 15) == 0) {           // lane 0 -> rows 0..7, lane 16 -> rows 8..15
        float s = 0.0f;
#pragma unroll
        for (int m = 0; m < 8; ++m) {
            float x = cneg[m] - cpos[m];              // neg_score - pos_score
            s += fmaxf(x, 0.0f) + log1pf(expf(-fabsf(x)));   // stable softplus
        }
        atomicAdd(mf_accum, s);
    }
}

// ------------------------------------------------------------- L2 reg loss
__global__ void k_emb_loss(const float* __restrict__ ue, const float* __restrict__ ie,
                           const int* __restrict__ users,
                           const int* __restrict__ pos_items,
                           const int* __restrict__ neg_items,
                           float* __restrict__ sq_accum) {
    __shared__ float red[256];
    int i = blockIdx.x * blockDim.x + threadIdx.x;    // 3*BATCH*16 exactly
    float s = 0.0f;
    if (i < 3 * BATCH * 16) {
        int c   = i & 15;
        int row = i >> 4;
        int b   = row % BATCH;
        int sel = row / BATCH;
        const float* base;
        if (sel == 0)      base = ue + (size_t)users[b] * EMB_DIM;
        else if (sel == 1) base = ie + (size_t)pos_items[b] * EMB_DIM;
        else               base = ie + (size_t)neg_items[b] * EMB_DIM;
        float4 v = *(const float4*)&base[c * 4];
        s = v.x * v.x + v.y * v.y + v.z * v.z + v.w * v.w;
    }
    red[threadIdx.x] = s;
    __syncthreads();
    for (int st = 128; st > 0; st >>= 1) {
        if ((int)threadIdx.x < st) red[threadIdx.x] += red[threadIdx.x + st];
        __syncthreads();
    }
    if (threadIdx.x == 0) atomicAdd(sq_accum, red[0]);
}

__global__ void k_final(const float* __restrict__ loss2, float* __restrict__ out) {
    if (threadIdx.x == 0 && blockIdx.x == 0) {
        out[0] = loss2[0] * (1.0f / (float)BATCH);
        out[1] = EMB_REG * loss2[1];
    }
}

// ------------------------------------------------------------------ driver
extern "C" void kernel_launch(void* const* d_in, const int* in_sizes, int n_in,
                              void* d_out, int out_size, void* d_ws, size_t ws_size,
                              hipStream_t stream) {
    (void)in_sizes; (void)n_in; (void)out_size; (void)ws_size;
    const float* ue    = (const float*)d_in[0];
    const float* ie    = (const float*)d_in[1];
    const int*   all_h = (const int*)d_in[2];
    const int*   all_t = (const int*)d_in[3];
    const int*   users = (const int*)d_in[4];
    const int*   pos   = (const int*)d_in[5];
    const int*   neg   = (const int*)d_in[6];

    // workspace layout (floats): deg | dinv | bufA | bufB | acc | loss2[2]
    float* ws    = (float*)d_ws;
    float* deg   = ws;
    float* dinv  = deg  + N_NODES;
    float* bufA  = dinv + N_NODES;
    float* bufB  = bufA + (size_t)N_NODES * EMB_DIM;
    float* acc   = bufB + (size_t)N_NODES * EMB_DIM;
    float* loss2 = acc  + (size_t)N_NODES * EMB_DIM;

    const int THREADS = 256;
    const int n4      = N_NODES * 16;                  // float4 count for tables

    k_zero<<<(N_NODES + THREADS - 1) / THREADS, THREADS, 0, stream>>>(deg, loss2);
    k_init<<<(n4 + THREADS - 1) / THREADS, THREADS, 0, stream>>>(
        ue, ie, (float4*)bufA, (float4*)acc);
    k_deg<<<(N_EDGES + THREADS - 1) / THREADS, THREADS, 0, stream>>>(all_h, deg);
    k_dinv<<<(N_NODES + THREADS - 1) / THREADS, THREADS, 0, stream>>>(deg, dinv);

    float* cur = bufA;
    float* nxt = bufB;
    const int spmm_blocks = (int)(((long)N_EDGES * 16 + THREADS - 1) / THREADS);
    for (int layer = 0; layer < N_LAYERS; ++layer) {
        k_copy<<<(n4 + THREADS - 1) / THREADS, THREADS, 0, stream>>>(
            (const float4*)cur, (float4*)nxt, n4);                  // residual: nxt = cur
        k_spmm<<<spmm_blocks, THREADS, 0, stream>>>(all_h, all_t, dinv, cur, nxt);
        k_accum<<<(n4 + THREADS - 1) / THREADS, THREADS, 0, stream>>>(
            (float4*)acc, (const float4*)nxt, n4);                  // acc += nxt
        float* tmp = cur; cur = nxt; nxt = tmp;                     // cur = new emb
    }

    // 512 tiles of 16 rows, 8 waves per 256-thread block -> 64 blocks
    k_mf_loss<<<(BATCH / 16) / 8, 256, 0, stream>>>(acc, users, pos, neg, loss2);
    k_emb_loss<<<(3 * BATCH * 16) / 256, 256, 0, stream>>>(
        ue, ie, users, pos, neg, loss2 + 1);
    k_final<<<1, 1, 0, stream>>>(loss2, (float*)d_out);
}
